// Swin_31293131718958
// MI455X (gfx1250) — matrix operand order repaired and checked by reading, not compiled, not run on hardware
//
#include <hip/hip_runtime.h>

// ---------------------------------------------------------------------------
// Swin-T forward for MI455X (gfx1250, wave32). All GEMMs use
// v_wmma_f32_16x16x32_f16 (f16 in, f32 accumulate). Residual stream f32.
// ---------------------------------------------------------------------------

typedef __attribute__((ext_vector_type(16))) _Float16 v16h;
typedef __attribute__((ext_vector_type(8)))  float    v8f;

#define WMMA_F16(a, b, c) \
  __builtin_amdgcn_wmma_f32_16x16x32_f16(false, (a), false, (b), (short)0, (c), false, false)

// ---- WMMA fragment loaders (contiguous k = 16*(lane/16) + s order, wave32) -------------

// A fragment, 16x32 f16, from row-major [.., ld] at (row0, k0).
// lane<16: K k0..k0+15; lane>=16: K k0+16..k0+31.  Two 16B loads.
static __device__ __forceinline__ v16h frag_a_rm(const _Float16* p, int ld,
                                                 int row0, int k0) {
  int lane = threadIdx.x & 31;
  const _Float16* s = p + (size_t)(row0 + (lane & 15)) * ld + k0 + ((lane >> 4) << 4);
  union { v16h v; uint4 u[2]; } r;
  r.u[0] = *(const uint4*)(s);
  r.u[1] = *(const uint4*)(s + 8);
  return r.v;
}

// B fragment, 32x16 f16, where B = W^T and W is row-major [N, K] (ld = K):
// B(k, n) = W[n][k]. lane<16 holds K k0..k0+15 of column n0+(lane&15).
static __device__ __forceinline__ v16h frag_b_wt(const _Float16* W, int ld,
                                                 int n0, int k0) {
  int lane = threadIdx.x & 31;
  const _Float16* s = W + (size_t)(n0 + (lane & 15)) * ld + k0 + ((lane >> 4) << 4);
  union { v16h v; uint4 u[2]; } r;
  r.u[0] = *(const uint4*)(s);
  r.u[1] = *(const uint4*)(s + 8);
  return r.v;
}

// B fragment, 32x16 f16, from row-major V [K, ld] (B(k,n) = V[k][n0+n]).
static __device__ __forceinline__ v16h frag_b_kn(const _Float16* V, int ld,
                                                 int k0, int n0) {
  int lane = threadIdx.x & 31;
  int n = n0 + (lane & 15);
  int kb = k0 + ((lane >> 4) << 4);
  v16h r;
#pragma unroll
  for (int j = 0; j < 16; ++j) r[j] = V[(size_t)(kb + j) * ld + n];
  return r;
}

// ---- Generic GEMM: C[M,N] = A[M,K](f16) * W[N,K]^T(f16) + bias ------------
// Block = 128 threads = 4 waves; each wave one 16x16 tile; block tile 32x32.
// M, N multiples of 32; K multiple of 32.
__global__ __launch_bounds__(128) void gemm_wmma(
    const _Float16* __restrict__ A, const _Float16* __restrict__ W,
    const float* __restrict__ bias, float* __restrict__ Cout,
    int M, int N, int K) {
  (void)M;
  int warp = threadIdx.x >> 5, lane = threadIdx.x & 31;
  int m0 = blockIdx.y * 32 + ((warp >> 1) << 4);
  int n0 = blockIdx.x * 32 + ((warp & 1) << 4);
  v8f acc = {};
  for (int k = 0; k < K; k += 32) {
    v16h a = frag_a_rm(A, K, m0, k);
    v16h b = frag_b_wt(W, K, n0, k);
    acc = WMMA_F16(a, b, acc);
  }
  int col = n0 + (lane & 15);
  int row = m0 + ((lane >> 4) << 3);
  float bv = bias ? bias[col] : 0.f;
#pragma unroll
  for (int r = 0; r < 8; ++r)
    Cout[(size_t)(row + r) * N + col] = acc[r] + bv;
}

// ---- Fused window attention (one block per (window, head)) ----------------
// 49 tokens padded to 64, head dim 32. S = qk^T (WMMA) + rpb (+shift mask),
// softmax over 49 keys, O = S*V (WMMA).  o written as f16 [nwin*49, C].
__global__ __launch_bounds__(256) void attn_kernel(
    const float* __restrict__ qkv,  // [nwin*49, 3C]
    const float* __restrict__ rpb,  // [169, nh]
    _Float16* __restrict__ o,       // [nwin*49, C]
    int C, int nh, int nW, int res, int shift) {
  __shared__ _Float16 qs[64 * 32], ks[64 * 32], vs[64 * 32];
  __shared__ float    sf[64 * 64];
  __shared__ _Float16 s16[64 * 64];
  int bw = blockIdx.x, h = blockIdx.y, tid = threadIdx.x;
  int C3 = 3 * C;
  for (int i = tid; i < 64 * 32; i += 256) {
    int t = i >> 5, c = i & 31;
    float qv = 0.f, kv = 0.f, vv = 0.f;
    if (t < 49) {
      size_t ro = (size_t)(bw * 49 + t) * C3 + h * 32 + c;
      qv = qkv[ro] * 0.17677669529663687f;  // 32^-0.5
      kv = qkv[ro + C];
      vv = qkv[ro + 2 * C];
    }
    qs[i] = (_Float16)qv; ks[i] = (_Float16)kv; vs[i] = (_Float16)vv;
  }
  __syncthreads();
  int warp = tid >> 5, lane = tid & 31;
  int wimg = bw % (nW * nW);
  int wy = wimg / nW, wx = wimg % nW;
  // S = q * k^T : 16 WMMA tiles over 8 waves
  for (int tile = warp; tile < 16; tile += 8) {
    int tm = tile >> 2, tn = tile & 3;
    v16h a = frag_a_rm(qs, 32, tm * 16, 0);
    v16h b = frag_b_wt(ks, 32, tn * 16, 0);
    v8f c = {};
    c = WMMA_F16(a, b, c);
    int n = tn * 16 + (lane & 15);
    int mb = tm * 16 + ((lane >> 4) << 3);
#pragma unroll
    for (int r = 0; r < 8; ++r) {
      int m = mb + r;  // m = query token, n = key token
      float val;
      if (m < 49 && n < 49) {
        val = c[r];
        int idx = (m / 7 - n / 7 + 6) * 13 + (m % 7 - n % 7 + 6);
        val += rpb[idx * nh + h];
        if (shift > 0) {
          int ym = wy * 7 + m / 7, xm = wx * 7 + m % 7;
          int yn = wy * 7 + n / 7, xn = wx * 7 + n % 7;
          int gm = (ym < res - 7 ? 0 : (ym < res - 3 ? 1 : 2)) * 3 +
                   (xm < res - 7 ? 0 : (xm < res - 3 ? 1 : 2));
          int gn = (yn < res - 7 ? 0 : (yn < res - 3 ? 1 : 2)) * 3 +
                   (xn < res - 7 ? 0 : (xn < res - 3 ? 1 : 2));
          if (gm != gn) val -= 100.f;
        }
      } else {
        val = -1e30f;
      }
      sf[m * 64 + n] = val;
    }
  }
  __syncthreads();
  if (tid < 64) {
    int m = tid;
    if (m < 49) {
      float mx = -1e30f;
      for (int n = 0; n < 49; ++n) mx = fmaxf(mx, sf[m * 64 + n]);
      float sum = 0.f;
      for (int n = 0; n < 49; ++n) sum += __expf(sf[m * 64 + n] - mx);
      float inv = 1.f / sum;
      for (int n = 0; n < 64; ++n)
        s16[m * 64 + n] = (n < 49) ? (_Float16)(__expf(sf[m * 64 + n] - mx) * inv)
                                   : (_Float16)0.f;
    } else {
      for (int n = 0; n < 64; ++n) s16[m * 64 + n] = (_Float16)0.f;
    }
  }
  __syncthreads();
  {  // O = S * V : 8 WMMA tiles, one per wave, K = 64 (two steps)
    int tm = warp >> 1, tn = warp & 1;
    v8f c = {};
    for (int k = 0; k < 64; k += 32) {
      v16h a = frag_a_rm(s16, 64, tm * 16, k);
      v16h b = frag_b_kn(vs, 32, k, tn * 16);
      c = WMMA_F16(a, b, c);
    }
    int n = tn * 16 + (lane & 15);
    int mb = tm * 16 + ((lane >> 4) << 3);
#pragma unroll
    for (int r = 0; r < 8; ++r) {
      int m = mb + r;
      if (m < 49)
        o[(size_t)(bw * 49 + m) * C + h * 32 + n] = (_Float16)c[r];
    }
  }
}

// ---- LayerNorm (one block per row), OUT = float or _Float16 ---------------
template <typename OUT>
__global__ __launch_bounds__(256) void ln_kernel(
    const float* __restrict__ x, const float* __restrict__ g,
    const float* __restrict__ b, OUT* __restrict__ y, int C) {
  __shared__ float red[256];
  int row = blockIdx.x;
  const float* xr = x + (size_t)row * C;
  float s = 0.f;
  for (int c = threadIdx.x; c < C; c += 256) s += xr[c];
  red[threadIdx.x] = s;
  __syncthreads();
  for (int o = 128; o > 0; o >>= 1) {
    if (threadIdx.x < o) red[threadIdx.x] += red[threadIdx.x + o];
    __syncthreads();
  }
  float mean = red[0] / C;
  __syncthreads();
  float v = 0.f;
  for (int c = threadIdx.x; c < C; c += 256) { float d = xr[c] - mean; v += d * d; }
  red[threadIdx.x] = v;
  __syncthreads();
  for (int o = 128; o > 0; o >>= 1) {
    if (threadIdx.x < o) red[threadIdx.x] += red[threadIdx.x + o];
    __syncthreads();
  }
  float inv = rsqrtf(red[0] / C + 1e-5f);
  OUT* yr = y + (size_t)row * C;
  for (int c = threadIdx.x; c < C; c += 256)
    yr[c] = (OUT)((xr[c] - mean) * inv * g[c] + b[c]);
}

// ---- window partition (roll by -shift folded into gather) -----------------
__global__ __launch_bounds__(256) void win_gather(
    const _Float16* __restrict__ y, _Float16* __restrict__ xw,
    int res, int C, int shift, long total) {
  long i = (long)blockIdx.x * 256 + threadIdx.x;
  if (i >= total) return;
  int nW = res / 7;
  int c = (int)(i % C);
  long t = i / C;
  int p = (int)(t % 49);
  long w = t / 49;
  int wimg = (int)(w % (nW * nW));
  int b = (int)(w / (nW * nW));
  int sy = (wimg / nW) * 7 + p / 7 + shift; if (sy >= res) sy -= res;
  int sx = (wimg % nW) * 7 + p % 7 + shift; if (sx >= res) sx -= res;
  xw[i] = y[((size_t)b * res * res + (size_t)sy * res + sx) * C + c];
}

// inverse (window -> image, roll back) with residual add; mapping is bijective
__global__ __launch_bounds__(256) void win_scatter_add(
    const float* __restrict__ t, float* __restrict__ x,
    int res, int C, int shift, long total) {
  long i = (long)blockIdx.x * 256 + threadIdx.x;
  if (i >= total) return;
  int nW = res / 7;
  int c = (int)(i % C);
  long tt = i / C;
  int p = (int)(tt % 49);
  long w = tt / 49;
  int wimg = (int)(w % (nW * nW));
  int b = (int)(w / (nW * nW));
  int sy = (wimg / nW) * 7 + p / 7 + shift; if (sy >= res) sy -= res;
  int sx = (wimg % nW) * 7 + p % 7 + shift; if (sx >= res) sx -= res;
  x[((size_t)b * res * res + (size_t)sy * res + sx) * C + c] += t[i];
}

// ---- patch-merge 2x2 concat gather ----------------------------------------
__global__ __launch_bounds__(256) void merge_gather(
    const float* __restrict__ x, float* __restrict__ o,
    int res, int C, long total) {
  long i = (long)blockIdx.x * 256 + threadIdx.x;
  if (i >= total) return;
  int C4 = 4 * C;
  int cc = (int)(i % C4);
  long t = i / C4;
  int g = cc / C, c = cc % C;
  int h2 = res / 2;
  int tx = (int)(t % h2); t /= h2;
  int ty = (int)(t % h2);
  int b = (int)(t / h2);
  int dy = g & 1, dx = g >> 1;  // order: (0,0),(1,0),(0,1),(1,1)
  o[i] = x[((size_t)b * res * res + (size_t)(2 * ty + dy) * res + (2 * tx + dx)) * C + c];
}

// ---- preprocess: *0.5+0.5, 2x2 area pool 448->224, normalize --------------
__global__ __launch_bounds__(256) void preprocess(
    const float* __restrict__ x, float* __restrict__ pre, long total) {
  long i = (long)blockIdx.x * 256 + threadIdx.x;
  if (i >= total) return;
  int px = (int)(i % 224); long t = i / 224;
  int py = (int)(t % 224); t /= 224;
  int c = (int)(t % 3);
  int b = (int)(t / 3);
  const float mean[3] = {0.48145466f, 0.4578275f, 0.40821073f};
  const float istd[3] = {1.f / 0.229f, 1.f / 0.224f, 1.f / 0.225f};
  size_t base = (((size_t)b * 3 + c) * 448 + 2 * py) * 448 + 2 * px;
  float s = x[base] + x[base + 1] + x[base + 448] + x[base + 449];
  s = s * 0.125f + 0.5f;  // (mean of 4) * 0.5 + 0.5
  pre[i] = (s - mean[c]) * istd[c];
}

// ---- patch embed: conv 4x4 stride 4 (K=48, VALU kernel) -------------------
__global__ __launch_bounds__(256) void patch_embed(
    const float* __restrict__ pre, const float* __restrict__ w,
    const float* __restrict__ bias, float* __restrict__ out, long total) {
  long i = (long)blockIdx.x * 256 + threadIdx.x;
  if (i >= total) return;
  int cout = (int)(i % 96);
  long t = i / 96;
  int tx = (int)(t % 56); t /= 56;
  int ty = (int)(t % 56);
  int b = (int)(t / 56);
  float s = bias[cout];
#pragma unroll
  for (int ci = 0; ci < 3; ++ci)
#pragma unroll
    for (int ky = 0; ky < 4; ++ky)
#pragma unroll
      for (int kx = 0; kx < 4; ++kx)
        s += w[((cout * 3 + ci) * 4 + ky) * 4 + kx] *
             pre[(((size_t)b * 3 + ci) * 224 + ty * 4 + ky) * 224 + tx * 4 + kx];
  out[i] = s;
}

// ---- misc elementwise ------------------------------------------------------
__global__ __launch_bounds__(256) void gelu_f16(const float* __restrict__ a,
                                                _Float16* __restrict__ o, long n) {
  long i = (long)blockIdx.x * 256 + threadIdx.x;
  if (i >= n) return;
  float v = a[i];
  o[i] = (_Float16)(0.5f * v * (1.f + erff(v * 0.70710678118654752f)));
}

__global__ __launch_bounds__(256) void cvt_f16(const float* __restrict__ a,
                                               _Float16* __restrict__ o, int n) {
  int i = blockIdx.x * 256 + threadIdx.x;
  if (i < n) o[i] = (_Float16)a[i];
}

__global__ __launch_bounds__(256) void add_inplace(float* __restrict__ x,
                                                   const float* __restrict__ t, long n) {
  long i = (long)blockIdx.x * 256 + threadIdx.x;
  if (i < n) x[i] += t[i];
}

__global__ __launch_bounds__(256) void final_mean(const float* __restrict__ t,
                                                  float* __restrict__ out,
                                                  int B, int L, int C) {
  int i = blockIdx.x * 256 + threadIdx.x;
  if (i >= B * C) return;
  int c = i % C, b = i / C;
  float s = 0.f;
  for (int l = 0; l < L; ++l) s += t[((size_t)b * L + l) * C + c];
  out[i] = s * (1.f / L);
}

// ---------------------------------------------------------------------------
// Host orchestration
// ---------------------------------------------------------------------------
struct BlockP {
  const float *fc1_b, *fc1_w, *fc2_b, *fc2_w, *n1b, *n1g, *n2b, *n2g;
  const float *proj_b, *proj_w, *qkv_b, *qkv_w, *rpb;
};
struct DsP { const float *nb, *ng, *red_w; };

extern "C" void kernel_launch(void* const* d_in, const int* in_sizes, int n_in,
                              void* d_out, int out_size, void* d_ws, size_t ws_size,
                              hipStream_t stream) {
  (void)out_size;
  static const int DIMS[4]  = {96, 192, 384, 768};
  static const int DEPTH[4] = {2, 2, 6, 2};
  static const int HEADS[4] = {3, 6, 12, 24};
  const int B = 32;

  // ---- locate x (32*3*448*448 f32) vs params; params leaves in JAX sorted-
  // key order: norm{b,g}, patch_embed{b,nb,ng,w}, stages[i]{blocks[j]{...}, downsample{nb,ng,red_w}}
  const float* x_in;
  int pi;
  if (in_sizes[0] == 32 * 3 * 448 * 448) { x_in = (const float*)d_in[0]; pi = 1; }
  else { x_in = (const float*)d_in[n_in - 1]; pi = 0; }
  auto next = [&]() { return (const float*)d_in[pi++]; };

  const float* norm_b = next(); const float* norm_g = next();
  const float* pe_b = next();   const float* pe_nb = next();
  const float* pe_ng = next();  const float* pe_w = next();
  BlockP blk[12]; DsP dsp[3];
  {
    int bi = 0;
    for (int i = 0; i < 4; ++i) {
      for (int j = 0; j < DEPTH[i]; ++j) {
        BlockP& p = blk[bi++];
        p.fc1_b = next(); p.fc1_w = next(); p.fc2_b = next(); p.fc2_w = next();
        p.n1b = next();   p.n1g = next();   p.n2b = next();   p.n2g = next();
        p.proj_b = next(); p.proj_w = next(); p.qkv_b = next(); p.qkv_w = next();
        p.rpb = next();
      }
      if (i < 3) { dsp[i].nb = next(); dsp[i].ng = next(); dsp[i].red_w = next(); }
    }
  }

  // ---- workspace carve ----
  const size_t MAXTC = 9633792;    // max tokens*C   (100352 * 96)
  const size_t MAX4C = 38535168;   // max tokens*4C  (100352 * 384)
  char* ws = (char*)d_ws;
  size_t off = 0;
  auto carve = [&](size_t bytes) -> void* {
    void* p = ws + off;
    off += (bytes + 255) & ~(size_t)255;
    return p;
  };
  float*     pre  = (float*)carve((size_t)B * 3 * 224 * 224 * 4);
  float*     xbuf = (float*)carve(MAXTC * 4);      // residual stream (f32)
  float*     tmpC = (float*)carve(MAXTC * 4);      // proj-out / merge / final-LN
  _Float16*  y16  = (_Float16*)carve(MAXTC * 2);   // LN outputs (f16)
  _Float16*  xw16 = (_Float16*)carve(MAXTC * 2);   // windowed tokens (f16)
  _Float16*  o16  = (_Float16*)carve(MAXTC * 2);   // attention out (f16)
  float*     big  = (float*)carve(MAX4C * 4);      // qkv-out / fc1-out (f32)
  _Float16*  h16b = (_Float16*)carve(MAX4C * 2);   // gelu out (f16)
  _Float16 *wq[12], *wp[12], *w1[12], *w2[12], *wr[3];
  {
    int bi = 0;
    for (int i = 0; i < 4; ++i) {
      size_t C = DIMS[i];
      for (int j = 0; j < DEPTH[i]; ++j, ++bi) {
        wq[bi] = (_Float16*)carve(3 * C * C * 2);
        wp[bi] = (_Float16*)carve(C * C * 2);
        w1[bi] = (_Float16*)carve(4 * C * C * 2);
        w2[bi] = (_Float16*)carve(4 * C * C * 2);
      }
      if (i < 3) wr[i] = (_Float16*)carve(8 * C * C * 2);
    }
  }
  if (off > ws_size) return;  // workspace too small; nothing safe to do

  auto cvt = [&](const float* src, _Float16* dst, int n) {
    cvt_f16<<<dim3((n + 255) / 256), dim3(256), 0, stream>>>(src, dst, n);
  };
  // one-time (per call) f16 weight conversion
  {
    int bi = 0;
    for (int i = 0; i < 4; ++i) {
      int C = DIMS[i];
      for (int j = 0; j < DEPTH[i]; ++j, ++bi) {
        cvt(blk[bi].qkv_w, wq[bi], 3 * C * C);
        cvt(blk[bi].proj_w, wp[bi], C * C);
        cvt(blk[bi].fc1_w, w1[bi], 4 * C * C);
        cvt(blk[bi].fc2_w, w2[bi], 4 * C * C);
      }
      if (i < 3) cvt(dsp[i].red_w, wr[i], 8 * C * C);
    }
  }

  auto gemm = [&](const _Float16* A, const _Float16* W, const float* bias,
                  float* Cptr, int M, int N, int K) {
    gemm_wmma<<<dim3(N / 32, M / 32), dim3(128), 0, stream>>>(A, W, bias, Cptr, M, N, K);
  };

  // ---- stem ----
  {
    long tot = (long)B * 3 * 224 * 224;
    preprocess<<<dim3((tot + 255) / 256), dim3(256), 0, stream>>>(x_in, pre, tot);
    long tot2 = (long)B * 3136 * 96;
    patch_embed<<<dim3((tot2 + 255) / 256), dim3(256), 0, stream>>>(pre, pe_w, pe_b, xbuf, tot2);
    ln_kernel<float><<<dim3(B * 3136), dim3(256), 0, stream>>>(xbuf, pe_ng, pe_nb, xbuf, 96);
  }

  // ---- stages ----
  int res = 56;
  int Ti = B * res * res;  // tokens
  int bi = 0;
  for (int i = 0; i < 4; ++i) {
    int C = DIMS[i], nh = HEADS[i];
    int nW = res / 7, nwin = B * nW * nW;
    for (int j = 0; j < DEPTH[i]; ++j, ++bi) {
      const BlockP& p = blk[bi];
      int sh = (j % 2 == 1 && res > 7) ? 3 : 0;
      long tot = (long)Ti * C;
      long tot4 = (long)Ti * 4 * C;
      // attention half
      ln_kernel<_Float16><<<dim3(Ti), dim3(256), 0, stream>>>(xbuf, p.n1g, p.n1b, y16, C);
      win_gather<<<dim3((tot + 255) / 256), dim3(256), 0, stream>>>(y16, xw16, res, C, sh, tot);
      gemm(xw16, wq[bi], p.qkv_b, big, Ti, 3 * C, C);
      attn_kernel<<<dim3(nwin, nh), dim3(256), 0, stream>>>(big, p.rpb, o16, C, nh, nW, res, sh);
      gemm(o16, wp[bi], p.proj_b, tmpC, Ti, C, C);
      win_scatter_add<<<dim3((tot + 255) / 256), dim3(256), 0, stream>>>(tmpC, xbuf, res, C, sh, tot);
      // MLP half
      ln_kernel<_Float16><<<dim3(Ti), dim3(256), 0, stream>>>(xbuf, p.n2g, p.n2b, y16, C);
      gemm(y16, w1[bi], p.fc1_b, big, Ti, 4 * C, C);
      gelu_f16<<<dim3((tot4 + 255) / 256), dim3(256), 0, stream>>>(big, h16b, tot4);
      gemm(h16b, w2[bi], p.fc2_b, tmpC, Ti, C, 4 * C);
      add_inplace<<<dim3((tot + 255) / 256), dim3(256), 0, stream>>>(xbuf, tmpC, tot);
    }
    if (i < 3) {
      long tot = (long)Ti * C;  // == (Ti/4) * 4C
      merge_gather<<<dim3((tot + 255) / 256), dim3(256), 0, stream>>>(xbuf, tmpC, res, C, tot);
      ln_kernel<_Float16><<<dim3(Ti / 4), dim3(256), 0, stream>>>(tmpC, dsp[i].ng, dsp[i].nb, y16, 4 * C);
      gemm(y16, wr[i], nullptr, xbuf, Ti / 4, 2 * C, 4 * C);
      res /= 2;
      Ti /= 4;
    }
  }

  // ---- head: final LN + token mean ----
  ln_kernel<float><<<dim3(Ti), dim3(256), 0, stream>>>(xbuf, norm_g, norm_b, tmpC, 768);
  final_mean<<<dim3((B * 768 + 255) / 256), dim3(256), 0, stream>>>(tmpC, (float*)d_out, B, 49, 768);
}
